// TNNet_67654324847611
// MI455X (gfx1250) — matrix operand order repaired and checked by reading, not compiled, not run on hardware
//
#include <hip/hip_runtime.h>
#include <math.h>

typedef _Float16 half16 __attribute__((ext_vector_type(16)));
typedef _Float16 half8  __attribute__((ext_vector_type(8)));
typedef float    f32x8  __attribute__((ext_vector_type(8)));
typedef int      v4i    __attribute__((ext_vector_type(4)));

#define EPS_LOG 1e-36f

// ---------------- CDNA5 async LDS copy (guarded) ----------------
#if __has_builtin(__builtin_amdgcn_global_load_async_to_lds_b128)
#define USE_ASYNC_LDS 1
#if __has_builtin(__builtin_amdgcn_s_wait_asynccnt)
#define WAIT_ASYNC(n) __builtin_amdgcn_s_wait_asynccnt(n)
#else
#define WAIT_ASYNC(n) asm volatile("s_wait_asynccnt %0" ::"n"(n) : "memory")
#endif
#else
#define USE_ASYNC_LDS 0
#define WAIT_ASYNC(n) ((void)0)
#endif

typedef v4i __attribute__((address_space(1))) as1_v4i;
typedef v4i __attribute__((address_space(3))) as3_v4i;

// Copy one 32KB slice (16384 f16) global->LDS; 8 x b128 per thread (256 threads).
__device__ __forceinline__ void issue_copy_slice(_Float16* dst, const _Float16* src,
                                                 int tid) {
#if USE_ASYNC_LDS
#pragma unroll
  for (int i = 0; i < 8; ++i) {
    int off = (i * 256 + tid) * 8;  // halves, 16B granules
    __builtin_amdgcn_global_load_async_to_lds_b128(
        (as1_v4i*)(size_t)(src + off),
        (as3_v4i*)(unsigned int)(size_t)(dst + off), 0, 0);
  }
#else
#pragma unroll
  for (int i = 0; i < 8; ++i) {
    int off = (i * 256 + tid) * 8;
    *(uint4*)(dst + off) = *(const uint4*)(src + off);
  }
#endif
}

// ---------------- helpers ----------------
__device__ __forceinline__ half16 h16zero() {
  half16 v;
#pragma unroll
  for (int i = 0; i < 16; ++i) v[i] = (_Float16)0.0f;
  return v;
}
__device__ __forceinline__ f32x8 f8zero() {
  f32x8 v;
#pragma unroll
  for (int i = 0; i < 8; ++i) v[i] = 0.0f;
  return v;
}
__device__ __forceinline__ f32x8 wmma16(half16 a, half16 b, f32x8 c) {
  return __builtin_amdgcn_wmma_f32_16x16x32_f16(false, a, false, b, (short)0, c,
                                                false, false);
}

// A operand: XT[u][k] row-major 32x32 f16; lane L<16: k {0..15}, else {16..31}.
__device__ __forceinline__ half16 load_A32(const _Float16* XT, int lane, int ut) {
  int u = ut * 16 + (lane & 15);
  const _Float16* row = XT + u * 32 + ((lane >> 4) ? 16 : 0);
  half8 lo = *(const half8*)(row);
  half8 hi = *(const half8*)(row + 8);
  half16 a;
#pragma unroll
  for (int i = 0; i < 8; ++i) { a[i] = lo[i]; a[i + 8] = hi[i]; }
  return a;
}
// B operand: YT[n][k] = B[k][n]; lane gets 32B contiguous k-run.
__device__ __forceinline__ half16 load_B32(const _Float16* YT, int lane, int nt) {
  int n = nt * 16 + (lane & 15);
  return *(const half16*)(YT + n * 32 + ((lane >> 4) ? 16 : 0));
}
// Store 16x16 f32 D tile transposed (scaled) into ZT[v][u] (32x32 f16).
__device__ __forceinline__ void store_DT(_Float16* ZT, int lane, int ut, int vt,
                                         f32x8 d, float s) {
  int v = vt * 16 + (lane & 15);
  _Float16* p = ZT + v * 32 + ut * 16 + ((lane >> 4) ? 8 : 0);
  half8 h;
#pragma unroll
  for (int i = 0; i < 8; ++i) h[i] = (_Float16)(d[i] * s);
  *(half8*)p = h;
}
// Embedding A tile: rows=samples(16), k=e (16 valid, top 16 zero-padded).
__device__ __forceinline__ half16 load_embA(const _Float16* embS, int lane) {
  const _Float16* p = embS + (lane & 15) * 16;
  half8 lo = *(const half8*)p;
  half8 hi = *(const half8*)(p + 8);
  half16 a;
#pragma unroll
  for (int i = 0; i < 8; ++i) { a[i] = lo[i]; a[i + 8] = hi[i]; }
  return a;
}
__device__ __forceinline__ float trig4(float xv, int i) {
  float a = ((i < 2) ? 1.5707963267948966f : 0.7853981633974483f) * xv;
  float s, c;
  __sincosf(a, &s, &c);
  return ((i & 1) ? c : s) * 0.7071067811865476f;
}

// ---------------- weight reorder/convert: f32 -> f16 blocks [l][r][e] ----------------
// wb16: 16 weights x 16 m -> 32KB blocks; ws16: 112 weights -> 32KB blocks.
__global__ __launch_bounds__(256) void tn_convert_kernel(
    const float* __restrict__ w_big, const float* __restrict__ w_small,
    _Float16* __restrict__ wb16, _Float16* __restrict__ ws16) {
  size_t t = (size_t)blockIdx.x * 256 + threadIdx.x;
  const size_t NB = (size_t)16 * 16 * 16384;
  const size_t NS = (size_t)112 * 16384;
  if (t < NB) {
    size_t blk = t >> 14;  // widx*16 + m
    int widx = (int)(blk >> 4), m = (int)(blk & 15);
    int rem = (int)(t & 16383);
    int l = rem >> 9, r = (rem >> 4) & 31, e = rem & 15;
    wb16[t] = (_Float16)w_big[(((size_t)(widx * 32 + l) * 16 + e) * 16 + m) * 32 + r];
  }
  if (t < NS) {
    size_t blk = t >> 14;  // small index
    int rem = (int)(t & 16383);
    int l = rem >> 9, r = (rem >> 4) & 31, e = rem & 15;
    ws16[t] = (_Float16)w_small[((size_t)(blk * 32 + l) * 16 + e) * 32 + r];
  }
}

// ---------------- per-sample logits kernel ----------------
// 64 blocks x 256 threads; 16 samples/block, 2 samples/wave (8 wave32s).
__global__ __launch_bounds__(256) void tn_logits_kernel(
    const float* __restrict__ x, const float* __restrict__ w_big,
    const _Float16* __restrict__ wb16, const _Float16* __restrict__ ws16,
    const float* __restrict__ w_small, float* __restrict__ out) {
  extern __shared__ char smem_raw[];
  _Float16* wslA = (_Float16*)(smem_raw);           // [32][32][16] W slice buf0
  _Float16* wslB = (_Float16*)(smem_raw + 32768);   // W slice buf1
  _Float16* psiT = (_Float16*)(smem_raw + 65536);   // [16][32][32] Psi^T per sample (r,l)
  _Float16* mt   = (_Float16*)(smem_raw + 98304);   // [16][32][32] M^T per sample (g,f)
  _Float16* gt   = (_Float16*)(smem_raw + 131072);  // [8][32][32]  per-wave G^T
  _Float16* embS = (_Float16*)(smem_raw + 147456);  // [16][16]
  float*    psis = (float*)(smem_raw + 147968);     // [8][32]

  const int tid = threadIdx.x;
  const int lane = tid & 31;
  const int wave = tid >> 5;
  const int n0 = blockIdx.x * 16;

  float lognorm0 = 0.0f, lognorm1 = 0.0f;
  f32x8 macc[2][2][2];

  auto stage_emb = [&](int site) {
    int s = tid >> 4, e = tid & 15;
    float x0 = x[((size_t)(n0 + s) * 128 + site) * 2 + 0];
    float x1 = x[((size_t)(n0 + s) * 128 + site) * 2 + 1];
    embS[(s << 4) + e] = (_Float16)(trig4(x0, e >> 2) * trig4(x1, e & 3));
  };

  auto site_end = [&]() {  // per-wave: max-abs rescale, M^T write, log accumulation
#pragma unroll
    for (int sl = 0; sl < 2; ++sl) {
      float mx = 0.0f;
#pragma unroll
      for (int ut = 0; ut < 2; ++ut)
#pragma unroll
        for (int vt = 0; vt < 2; ++vt)
#pragma unroll
          for (int i = 0; i < 8; ++i) mx = fmaxf(mx, fabsf(macc[sl][ut][vt][i]));
      for (int off = 16; off > 0; off >>= 1) mx = fmaxf(mx, __shfl_xor(mx, off, 32));
      if (sl == 0) lognorm0 += logf(mx + EPS_LOG);
      else         lognorm1 += logf(mx + EPS_LOG);
      float inv = 1.0f / mx;
      _Float16* mts = mt + (wave * 2 + sl) * 1024;
#pragma unroll
      for (int ut = 0; ut < 2; ++ut)
#pragma unroll
        for (int vt = 0; vt < 2; ++vt) store_DT(mts, lane, ut, vt, macc[sl][ut][vt], inv);
    }
  };

  // ============ site 0: Psi0[m,r] = sum_e emb_e * W0[0,e,m,r]; M1 = Psi0^T Psi0 ============
  for (int i = tid; i < 16 * 1024; i += 256) psiT[i] = (_Float16)0.0f;  // zero-pad K
  for (int i = tid; i < 8192; i += 256) {  // w_big[0][l=0][e][m][r] -> wslA[m][r][e]
    int mm = i >> 9, e = (i >> 5) & 15, r = i & 31;
    wslA[((((mm << 5) + r)) << 4) + e] = (_Float16)w_big[((e << 4) + mm) * 32 + r];
  }
  stage_emb(0);
  __syncthreads();
  {
    half16 aE = load_embA(embS, lane);
#pragma unroll
    for (int j = 0; j < 4; ++j) {  // 32 (m,ntile) pairs over 8 waves
      int p = wave * 4 + j;
      int mm = p >> 1, nt = p & 1;
      half16 b = (lane < 16)
                     ? *(const half16*)(wslA + (((mm << 5) + nt * 16 + lane) << 4))
                     : h16zero();
      f32x8 d = wmma16(aE, b, f8zero());
      int r = nt * 16 + (lane & 15);
      int sbase = (lane >> 4) * 8;
#pragma unroll
      for (int q = 0; q < 8; ++q)
        psiT[((((sbase + q) << 5) + r) << 5) + mm] = (_Float16)d[q];  // [s][r][m]
    }
  }
  __syncthreads();
#pragma unroll
  for (int sl = 0; sl < 2; ++sl) {  // M1 tiles (K=32, top half zeros)
    const _Float16* ps = psiT + (wave * 2 + sl) * 1024;
#pragma unroll
    for (int ut = 0; ut < 2; ++ut) {
      half16 a = load_A32(ps, lane, ut);
#pragma unroll
      for (int vt = 0; vt < 2; ++vt) {
        half16 b = load_B32(ps, lane, vt);
        macc[sl][ut][vt] = wmma16(a, b, f8zero());
      }
    }
  }
  site_end();

  // ============ sites 1..126 (double-buffered async W staging) ============
  for (int site = 1; site <= 126; ++site) {
    const bool big = (site & 7) == 0;
    const int md = big ? 16 : 1;
    const _Float16* wbase =
        big ? (wb16 + (((size_t)(site >> 3) * 16) << 14))
            : (ws16 + ((size_t)(site - (site >> 3) - 1) << 14));
    __syncthreads();
    stage_emb(site);
#pragma unroll
    for (int sl = 0; sl < 2; ++sl)
#pragma unroll
      for (int ut = 0; ut < 2; ++ut)
#pragma unroll
        for (int vt = 0; vt < 2; ++vt) macc[sl][ut][vt] = f8zero();

    _Float16* bufs[2] = {wslA, wslB};
    issue_copy_slice(bufs[0], wbase, tid);  // prefetch slice 0

    for (int m = 0; m < md; ++m) {
      if (m + 1 < md) {  // overlap: issue next slice, then wait for current only
        issue_copy_slice(bufs[(m + 1) & 1], wbase + ((size_t)(m + 1) << 14), tid);
        WAIT_ASYNC(8);
      } else {
        WAIT_ASYNC(0);
      }
      __syncthreads();
      const _Float16* wsl = bufs[m & 1];
      // --- Psi GEMM: 64 (l,ntile) tiles over 8 waves ---
      half16 aE = load_embA(embS, lane);
#pragma unroll
      for (int j = 0; j < 8; ++j) {
        int p = wave * 8 + j;
        int l = p >> 1, nt = p & 1;
        half16 b = (lane < 16)
                       ? *(const half16*)(wsl + (((l << 5) + nt * 16 + lane) << 4))
                       : h16zero();
        f32x8 d = wmma16(aE, b, f8zero());
        int r = nt * 16 + (lane & 15);
        int sbase = (lane >> 4) * 8;
#pragma unroll
        for (int q = 0; q < 8; ++q)
          psiT[((((sbase + q) << 5) + r) << 5) + l] = (_Float16)d[q];  // [s][r][l]
      }
      __syncthreads();
      // --- transfer: G = M^T Psi ; M' += G^T Psi (per wave, 2 samples) ---
#pragma unroll
      for (int sl = 0; sl < 2; ++sl) {
        int sm = wave * 2 + sl;
        const _Float16* mts = mt + sm * 1024;
        const _Float16* ps = psiT + sm * 1024;
        _Float16* g = gt + wave * 1024;
#pragma unroll
        for (int ut = 0; ut < 2; ++ut) {
          half16 a = load_A32(mts, lane, ut);
#pragma unroll
          for (int vt = 0; vt < 2; ++vt) {
            half16 b = load_B32(ps, lane, vt);
            f32x8 d = wmma16(a, b, f8zero());
            store_DT(g, lane, ut, vt, d, 1.0f);
          }
        }
#pragma unroll
        for (int ut = 0; ut < 2; ++ut) {
          half16 a = load_A32(g, lane, ut);
#pragma unroll
          for (int vt = 0; vt < 2; ++vt) {
            half16 b = load_B32(ps, lane, vt);
            macc[sl][ut][vt] = wmma16(a, b, macc[sl][ut][vt]);
          }
        }
      }
    }
    site_end();
  }

  // ============ site 127 (r=1): logit = log(psi^T M psi) + lognorm ============
  __syncthreads();
  stage_emb(127);
  __syncthreads();
  const float* wlast = w_small + (size_t)111 * (32 * 16 * 32);
#pragma unroll
  for (int sl = 0; sl < 2; ++sl) {
    int sm = wave * 2 + sl;
    float p = 0.0f;
#pragma unroll
    for (int e = 0; e < 16; ++e)
      p += (float)embS[(sm << 4) + e] * wlast[(((lane << 4) + e) << 5)];
    psis[(wave << 5) + lane] = p;  // wave-local share (LDS in-order per wave)
    float inner = 0.0f;
    const _Float16* mts = mt + sm * 1024;
#pragma unroll 4
    for (int f = 0; f < 32; ++f)
      inner += (float)mts[(lane << 5) + f] * psis[(wave << 5) + f];
    float part = p * inner;
    for (int off = 16; off > 0; off >>= 1) part += __shfl_xor(part, off, 32);
    if (lane == 0) out[n0 + sm] = logf(part + EPS_LOG) + (sl == 0 ? lognorm0 : lognorm1);
  }
}

// ---------------- log partition kernel (1 block) ----------------
__global__ __launch_bounds__(256) void tn_logz_kernel(const float* __restrict__ w_big,
                                                      const float* __restrict__ w_small,
                                                      float* __restrict__ out) {
  extern __shared__ char smem_raw[];
  float*    f0   = (float*)(smem_raw);             // 8192 f32 staging (site 0 / 127)
  _Float16* t1   = (_Float16*)(smem_raw);          // [8][32][32] em-slices (overlaps f0)
  _Float16* gtz  = (_Float16*)(smem_raw + 16384);  // [8][32][32] per-wave G^T
  _Float16* mtz  = (_Float16*)(smem_raw + 32768);  // [32][32] M^T
  float*    zsum = (float*)(smem_raw + 34816);     // [32][32] f32 M' accumulation
  float*    sc   = (float*)(smem_raw + 38912);     // scalars: lognorm, acc, wave maxes

  const int tid = threadIdx.x;
  const int lane = tid & 31;
  const int wave = tid >> 5;

  if (tid == 0) { sc[0] = 0.0f; sc[1] = 0.0f; }

  auto site_end_z = [&]() {
    float mx = 0.0f;
#pragma unroll
    for (int k = 0; k < 4; ++k) mx = fmaxf(mx, fabsf(zsum[tid * 4 + k]));
    for (int off = 16; off > 0; off >>= 1) mx = fmaxf(mx, __shfl_xor(mx, off, 32));
    if (lane == 0) sc[2 + wave] = mx;
    __syncthreads();
    if (tid == 0) {
      float m2 = 0.0f;
      for (int w = 0; w < 8; ++w) m2 = fmaxf(m2, sc[2 + w]);
      sc[10] = 1.0f / m2;
      sc[0] += logf(m2 + EPS_LOG);
    }
    __syncthreads();
    float inv = sc[10];
#pragma unroll
    for (int k = 0; k < 4; ++k) {
      int fg = tid * 4 + k, f = fg >> 5, g = fg & 31;
      mtz[(g << 5) + f] = (_Float16)(zsum[fg] * inv);
    }
    __syncthreads();
  };

  // site 0: M1[f,g] = sum_{em} A0[em,f] A0[em,g]
  for (int i = tid; i < 8192; i += 256) f0[i] = w_big[i];
  __syncthreads();
#pragma unroll
  for (int k = 0; k < 4; ++k) {
    int fg = tid * 4 + k, f = fg >> 5, g = fg & 31;
    float acc = 0.0f;
    for (int em = 0; em < 256; ++em) acc += f0[(em << 5) + f] * f0[(em << 5) + g];
    zsum[fg] = acc;
  }
  __syncthreads();
  site_end_z();

  f32x8 macc[2][2];
  for (int site = 1; site <= 126; ++site) {
    const bool big = (site & 7) == 0;
    const int md = big ? 16 : 1;
    const int nsl = md * 16;
    const float* wbase = big ? (w_big + (size_t)(site >> 3) * (32 * 16 * 16 * 32))
                             : (w_small + (size_t)(site - (site >> 3) - 1) * (32 * 16 * 32));
    __syncthreads();
    for (int i = tid; i < 1024; i += 256) zsum[i] = 0.0f;
#pragma unroll
    for (int ut = 0; ut < 2; ++ut)
#pragma unroll
      for (int vt = 0; vt < 2; ++vt) macc[ut][vt] = f8zero();
    const int nchunk = (nsl + 7) >> 3;
    for (int c = 0; c < nchunk; ++c) {
      __syncthreads();
      int base = c * 8;
      int cnt = nsl - base; if (cnt > 8) cnt = 8;
      for (int i = tid; i < (cnt << 10); i += 256) {  // A[l,e,m,r] -> t1[j][r][l]
        int j = i >> 10, rem = i & 1023, l = rem >> 5, r = rem & 31;
        int q = base + j;
        int e = q / md, m = q - e * md;
        t1[(((j << 5) + r) << 5) + l] =
            (_Float16)wbase[((size_t)((l << 4) + e) * md + m) * 32 + r];
      }
      __syncthreads();
      if (wave < cnt) {
        const _Float16* slc = t1 + wave * 1024;
        _Float16* g = gtz + wave * 1024;
#pragma unroll
        for (int ut = 0; ut < 2; ++ut) {
          half16 a = load_A32(mtz, lane, ut);
#pragma unroll
          for (int vt = 0; vt < 2; ++vt) {
            half16 b = load_B32(slc, lane, vt);
            f32x8 d = wmma16(a, b, f8zero());
            store_DT(g, lane, ut, vt, d, 1.0f);
          }
        }
#pragma unroll
        for (int ut = 0; ut < 2; ++ut) {
          half16 a = load_A32(g, lane, ut);
#pragma unroll
          for (int vt = 0; vt < 2; ++vt) {
            half16 b = load_B32(slc, lane, vt);
            macc[ut][vt] = wmma16(a, b, macc[ut][vt]);
          }
        }
      }
    }
    __syncthreads();
#pragma unroll
    for (int ut = 0; ut < 2; ++ut)
#pragma unroll
      for (int vt = 0; vt < 2; ++vt)
#pragma unroll
        for (int i = 0; i < 8; ++i) {
          int u = ut * 16 + (lane >> 4) * 8 + i;
          int v = vt * 16 + (lane & 15);
          atomicAdd(&zsum[(u << 5) + v], macc[ut][vt][i]);
        }
    __syncthreads();
    site_end_z();
  }

  // site 127: scalar = sum_{f,g} M[f,g] * sum_e A127[f,e] A127[g,e]
  __syncthreads();
  for (int i = tid; i < 512; i += 256)
    f0[i] = w_small[(size_t)111 * (32 * 16 * 32) + (size_t)i * 32];
  __syncthreads();
  float acc = 0.0f;
#pragma unroll
  for (int k = 0; k < 4; ++k) {
    int fg = tid * 4 + k, f = fg >> 5, g = fg & 31;
    float be = 0.0f;
#pragma unroll
    for (int e = 0; e < 16; ++e) be += f0[(f << 4) + e] * f0[(g << 4) + e];
    acc += be * (float)mtz[(g << 5) + f];
  }
  for (int off = 16; off > 0; off >>= 1) acc += __shfl_xor(acc, off, 32);
  if (lane == 0) atomicAdd(&sc[1], acc);
  __syncthreads();
  if (tid == 0) out[1024] = logf(sc[1] + EPS_LOG) + sc[0];
}

// ---------------- launcher ----------------
extern "C" void kernel_launch(void* const* d_in, const int* in_sizes, int n_in,
                              void* d_out, int out_size, void* d_ws, size_t ws_size,
                              hipStream_t stream) {
  (void)in_sizes; (void)n_in; (void)out_size; (void)ws_size;
  const float* x = (const float*)d_in[0];
  const float* w_big = (const float*)d_in[1];
  const float* w_small = (const float*)d_in[2];
  float* out = (float*)d_out;

  _Float16* wb16 = (_Float16*)d_ws;                      // 16*16 blocks of 32KB (8MB)
  _Float16* ws16 = wb16 + (size_t)16 * 16 * 16384;       // 112 blocks of 32KB (3.5MB)

  const int MAIN_SMEM = 148992;  // ~145.5 KB dynamic LDS (<< 320 KB/WGP)
  const int Z_SMEM = 40960;
  (void)hipFuncSetAttribute(reinterpret_cast<const void*>(tn_logits_kernel),
                            hipFuncAttributeMaxDynamicSharedMemorySize, MAIN_SMEM);
  (void)hipFuncSetAttribute(reinterpret_cast<const void*>(tn_logz_kernel),
                            hipFuncAttributeMaxDynamicSharedMemorySize, Z_SMEM);

  tn_convert_kernel<<<dim3(16384), dim3(256), 0, stream>>>(w_big, w_small, wb16, ws16);
  tn_logits_kernel<<<dim3(64), dim3(256), MAIN_SMEM, stream>>>(x, w_big, wb16, ws16,
                                                               w_small, out);
  tn_logz_kernel<<<dim3(1), dim3(256), Z_SMEM, stream>>>(w_big, w_small, out);
}
